// Decoder_88313117540686
// MI455X (gfx1250) — compile-verified, matrix operand order checked by reading, not run on hardware
//
#include <hip/hip_runtime.h>
#include <hip/hip_bf16.h>
#include <math.h>

// ---------------------------------------------------------------------------
// Problem constants (from reference): B=2048, T=64, HID=128, FEAT=128
// ---------------------------------------------------------------------------
#define BATCH 2048
#define TLEN  64
#define HID   128
#define FEAT  128

typedef __attribute__((ext_vector_type(16))) __bf16 v16bf;
typedef __attribute__((ext_vector_type(8)))  __bf16 v8bf;
typedef __attribute__((ext_vector_type(8)))  float  v8f;
typedef __attribute__((ext_vector_type(4)))  float  v4f;

// ---------------------------------------------------------------------------
// WMMA fragment loaders for v_wmma_f32_16x16x32_bf16 (wave32).
// 16-bit A/B layout (ISA 7.12.2): lane L(<16) holds row/col M=L with
// K = {k0..k0+7, k0+16..k0+23}, lane L+16 holds K shifted by +8.
// Caller passes k0 = kblock*32 + (lane>=16 ? 8 : 0); we fetch two contiguous
// 16-byte segments (b128 loads) and pack.
// ---------------------------------------------------------------------------
__device__ inline v16bf frag_from_bf16(const __bf16* row, int k0) {
  v8bf lo = *(const v8bf*)(row + k0);
  v8bf hi = *(const v8bf*)(row + k0 + 16);
  v16bf r;
#pragma unroll
  for (int i = 0; i < 8; ++i) { r[i] = lo[i]; r[i + 8] = hi[i]; }
  return r;
}

__device__ inline v16bf frag_from_f32(const float* row, int k0) {
  v4f a0 = *(const v4f*)(row + k0);
  v4f a1 = *(const v4f*)(row + k0 + 4);
  v4f b0 = *(const v4f*)(row + k0 + 16);
  v4f b1 = *(const v4f*)(row + k0 + 20);
  v16bf r;
#pragma unroll
  for (int i = 0; i < 4; ++i) {
    r[i]      = (__bf16)a0[i];
    r[4 + i]  = (__bf16)a1[i];
    r[8 + i]  = (__bf16)b0[i];
    r[12 + i] = (__bf16)b1[i];
  }
  return r;
}

__device__ inline v8f wmma_bf16(v16bf a, v16bf b, v8f c) {
  return __builtin_amdgcn_wmma_f32_16x16x32_bf16(false, a, false, b, (short)0, c,
                                                 false, false);
}

__device__ inline float sigm(float x) { return 1.0f / (1.0f + __expf(-x)); }

// ---------------------------------------------------------------------------
// Kernel 1: weight prep.
//  W1T[n][k] = bf16(attn_W1[k][n])  n<128, k<384 (transposed so B-fragments of
//              both attention GEMMs are contiguous in K)
//  WhhB[n][k] = bf16(W_hh[n][k])    (gates GEMM uses B[k][n]=W_hh[n][k], so
//              row-major W_hh already gives contiguous K per output column)
// ---------------------------------------------------------------------------
__global__ __launch_bounds__(256) void prep_kernel(const float* __restrict__ attn_W1,
                                                   const float* __restrict__ W_hh,
                                                   __bf16* __restrict__ W1T,
                                                   __bf16* __restrict__ WhhB) {
  int i = blockIdx.x * blockDim.x + threadIdx.x;
  const int n1 = FEAT * 384;          // 49152
  const int n2 = 4 * HID * HID;       // 65536
  if (i < n1) {
    int n = i / 384, k = i - n * 384;
    W1T[(size_t)n * 384 + k] = (__bf16)attn_W1[(size_t)k * FEAT + n];
  } else if (i < n1 + n2) {
    int j = i - n1;
    WhhB[j] = (__bf16)W_hh[j];
  }
}

// ---------------------------------------------------------------------------
// Kernel 2: time-invariant attention projection (hoisted out of the scan):
//   HWb[m][n] = sum_k H[m][k] * attn_W1[256+k][n] + attn_b1[n]
// m over B*T = 131072 rows. One block = one 16-row M tile, wave w = N-tile w.
// ---------------------------------------------------------------------------
__global__ __launch_bounds__(256) void hproj_kernel(const float* __restrict__ H,
                                                    const float* __restrict__ attn_b1,
                                                    const __bf16* __restrict__ W1T,
                                                    float* __restrict__ HWb) {
  const int tid = threadIdx.x;
  const int lane = tid & 31;
  const int wv = tid >> 5;            // 0..7 -> N tile
  const int half = lane >> 4;
  const int col = lane & 15;
  const size_t m0 = (size_t)blockIdx.x * 16;

  const float*  arow = H + (m0 + col) * HID;                       // A row M=col
  const __bf16* brow = W1T + (size_t)(wv * 16 + col) * 384 + 256;  // W1_H column

  v8f acc = {};
#pragma unroll
  for (int ks = 0; ks < 4; ++ks) {      // K = 128 = 4 x 32
    int k0 = ks * 32 + half * 8;
    v16bf a = frag_from_f32(arow, k0);
    v16bf b = frag_from_bf16(brow, k0);
    acc = wmma_bf16(a, b, acc);
  }
  const int n = wv * 16 + col;
  const float bias = attn_b1[n];
#pragma unroll
  for (int r = 0; r < 8; ++r)
    HWb[(m0 + r + half * 8) * FEAT + n] = acc[r] + bias;
}

// ---------------------------------------------------------------------------
// Kernel 3: the recurrent decoder. One block = 16 batch rows, loops t=0..63
// internally (batch rows are independent, so no inter-block sync needed).
// ---------------------------------------------------------------------------
__global__ __launch_bounds__(256) void decoder_kernel(
    const float* __restrict__ H, const float* __restrict__ Y,
    const float* __restrict__ attn_W2, const float* __restrict__ attn_b2,
    const float* __restrict__ W_ih, const float* __restrict__ b_ih,
    const float* __restrict__ b_hh, const float* __restrict__ fc_W,
    const float* __restrict__ fc_b, const float* __restrict__ fcf_W,
    const float* __restrict__ fcf_b, const float* __restrict__ HWb,
    const __bf16* __restrict__ W1T, const __bf16* __restrict__ WhhB,
    float* __restrict__ out) {
  // +8 bf16 (16B) row pad on s_dc so the 16 per-lane b128 A-fragment reads hit
  // distinct LDS bank groups.
  __shared__ __bf16 s_dc[16][264];   // [d | c] as bf16 (A operand of both GEMMs)
  __shared__ float  s_dcW[16][128];  // dc @ W1_dc
  __shared__ float  s_d[16][128];
  __shared__ float  s_c[16][128];
  __shared__ float  s_ctx[16][128];
  __shared__ float  s_e[16][64];     // attention logits -> softmax weights
  __shared__ float  s_w2[128];
  __shared__ float  s_yt[16];

  const int tid = threadIdx.x;
  const int lane = tid & 31;
  const int wv = tid >> 5;          // 0..7
  const int half = lane >> 4;
  const int col = lane & 15;
  const int b0 = blockIdx.x * 16;

  for (int i = tid; i < 16 * 128; i += 256) {
    (&s_d[0][0])[i] = 0.0f;
    (&s_c[0][0])[i] = 0.0f;
    (&s_ctx[0][0])[i] = 0.0f;
  }
  if (tid < 128) s_w2[tid] = attn_W2[tid];
  __syncthreads();

  const float b2  = attn_b2[0];
  const float fcb = fc_b[0];

  for (int t = 0; t < TLEN; ++t) {
    // (1) dc = cat(d, c) in bf16
    for (int i = tid; i < 16 * 256; i += 256) {
      int r = i >> 8, k = i & 255;
      float v = (k < 128) ? s_d[r][k] : s_c[r][k - 128];
      s_dc[r][k] = (__bf16)v;
    }
    __syncthreads();

    // (2) dcW = dc @ W1_dc   ([16,256]x[256,128]); wave wv -> N-tile wv
    {
      v8f acc = {};
      const __bf16* arow = &s_dc[col][0];
      const __bf16* brow = W1T + (size_t)(wv * 16 + col) * 384;  // k < 256 half
#pragma unroll
      for (int ks = 0; ks < 8; ++ks) {
        int k0 = ks * 32 + half * 8;
        v16bf a = frag_from_bf16(arow, k0);
        v16bf b = frag_from_bf16(brow, k0);
        acc = wmma_bf16(a, b, acc);
      }
      const int n = wv * 16 + col;
#pragma unroll
      for (int r = 0; r < 8; ++r) s_dcW[r + half * 8][n] = acc[r];
    }
    __syncthreads();

    // (3) logits e[b,tt] = tanh(HWb + dcW) . w2 + b2   (HWb is L2-resident)
    for (int p = tid; p < 16 * TLEN; p += 256) {
      int r = p >> 6, tt = p & 63;
      const float* hp = HWb + ((size_t)(b0 + r) * TLEN + tt) * FEAT;
      float acc = b2;
#pragma unroll 4
      for (int f = 0; f < FEAT; f += 4) {
        v4f hv = *(const v4f*)(hp + f);
        acc += tanhf(hv[0] + s_dcW[r][f + 0]) * s_w2[f + 0];
        acc += tanhf(hv[1] + s_dcW[r][f + 1]) * s_w2[f + 1];
        acc += tanhf(hv[2] + s_dcW[r][f + 2]) * s_w2[f + 2];
        acc += tanhf(hv[3] + s_dcW[r][f + 3]) * s_w2[f + 3];
      }
      s_e[r][tt] = acc;
    }
    __syncthreads();

    // (4) softmax over T per row
    if (tid < 16) {
      float m = s_e[tid][0];
      for (int tt = 1; tt < TLEN; ++tt) m = fmaxf(m, s_e[tid][tt]);
      float s = 0.0f;
      for (int tt = 0; tt < TLEN; ++tt) {
        float ex = __expf(s_e[tid][tt] - m);
        s_e[tid][tt] = ex;
        s += ex;
      }
      float inv = 1.0f / s;
      for (int tt = 0; tt < TLEN; ++tt) s_e[tid][tt] *= inv;
    }
    __syncthreads();

    // (5) context = sum_t beta * H   (thread owns (row, 8 features))
    {
      int r = tid >> 4, f0 = (tid & 15) * 8;
      v4f a0 = {}, a1 = {};
      const float* hp = H + (size_t)(b0 + r) * TLEN * HID + f0;
      for (int tt = 0; tt < TLEN; ++tt) {
        float bta = s_e[r][tt];
        a0 += bta * (*(const v4f*)(hp));
        a1 += bta * (*(const v4f*)(hp + 4));
        hp += HID;
      }
      *(v4f*)&s_ctx[r][f0] = a0;
      *(v4f*)&s_ctx[r][f0 + 4] = a1;
    }
    __syncthreads();

    // (6) y_tilde = fc(cat(context, y_t))
    if (tid < 16) {
      float acc = fcb + Y[(size_t)(b0 + tid) * TLEN + t] * fc_W[128];
      for (int f = 0; f < HID; ++f) acc += s_ctx[tid][f] * fc_W[f];
      s_yt[tid] = acc;
    }
    __syncthreads();

    // (7) gates = y_tilde*W_ih^T + d @ W_hh^T + b_ih + b_hh, then LSTM cell.
    // Wave wv owns hidden columns h = wv*16 + col for all 4 gates; bias and
    // y_tilde terms are folded into the WMMA C accumulator; pointwise LSTM
    // stays in accumulator registers.
    {
      const int h = wv * 16 + col;
      v8f aI, aF, aG, aO;
#pragma unroll
      for (int r = 0; r < 8; ++r) {
        int M = r + half * 8;
        float yt = s_yt[M];
        aI[r] = yt * W_ih[h]       + b_ih[h]       + b_hh[h];
        aF[r] = yt * W_ih[128 + h] + b_ih[128 + h] + b_hh[128 + h];
        aG[r] = yt * W_ih[256 + h] + b_ih[256 + h] + b_hh[256 + h];
        aO[r] = yt * W_ih[384 + h] + b_ih[384 + h] + b_hh[384 + h];
      }
      const __bf16* arow = &s_dc[col][0];   // d occupies cols 0..127 of dc
      const __bf16* bI = WhhB + (size_t)(h)       * HID;
      const __bf16* bF = WhhB + (size_t)(128 + h) * HID;
      const __bf16* bG = WhhB + (size_t)(256 + h) * HID;
      const __bf16* bO = WhhB + (size_t)(384 + h) * HID;
#pragma unroll
      for (int ks = 0; ks < 4; ++ks) {
        int k0 = ks * 32 + half * 8;
        v16bf a = frag_from_bf16(arow, k0);
        aI = wmma_bf16(a, frag_from_bf16(bI, k0), aI);
        aF = wmma_bf16(a, frag_from_bf16(bF, k0), aF);
        aG = wmma_bf16(a, frag_from_bf16(bG, k0), aG);
        aO = wmma_bf16(a, frag_from_bf16(bO, k0), aO);
      }
#pragma unroll
      for (int r = 0; r < 8; ++r) {
        int M = r + half * 8;
        float ig = sigm(aI[r]);
        float fg = sigm(aF[r]);
        float gg = tanhf(aG[r]);
        float og = sigm(aO[r]);
        float cn = fg * s_c[M][h] + ig * gg;
        s_c[M][h] = cn;
        s_d[M][h] = og * tanhf(cn);
      }
    }
    __syncthreads();
  }

  // final: y_pred = cat(d, context) @ fcf_W + fcf_b
  if (tid < 16) {
    float acc = fcf_b[0];
    for (int f = 0; f < HID; ++f)
      acc += s_d[tid][f] * fcf_W[f] + s_ctx[tid][f] * fcf_W[128 + f];
    out[b0 + tid] = acc;
  }
}

// ---------------------------------------------------------------------------
// Host side
// ---------------------------------------------------------------------------
extern "C" void kernel_launch(void* const* d_in, const int* in_sizes, int n_in,
                              void* d_out, int out_size, void* d_ws, size_t ws_size,
                              hipStream_t stream) {
  (void)in_sizes; (void)n_in; (void)out_size; (void)ws_size;
  const float* H       = (const float*)d_in[0];
  const float* Y       = (const float*)d_in[1];
  const float* attn_W1 = (const float*)d_in[2];
  const float* attn_b1 = (const float*)d_in[3];
  const float* attn_W2 = (const float*)d_in[4];
  const float* attn_b2 = (const float*)d_in[5];
  const float* W_ih    = (const float*)d_in[6];
  const float* W_hh    = (const float*)d_in[7];
  const float* b_ih    = (const float*)d_in[8];
  const float* b_hh    = (const float*)d_in[9];
  const float* fc_W    = (const float*)d_in[10];
  const float* fc_b    = (const float*)d_in[11];
  const float* fcf_W   = (const float*)d_in[12];
  const float* fcf_b   = (const float*)d_in[13];

  // workspace layout: [HWb f32 64MB][W1T bf16 96KB][WhhB bf16 128KB]
  float*  HWb  = (float*)d_ws;
  __bf16* W1T  = (__bf16*)((char*)d_ws + (size_t)BATCH * TLEN * FEAT * sizeof(float));
  __bf16* WhhB = W1T + (size_t)FEAT * 384;

  const int prep_n = FEAT * 384 + 4 * HID * HID;
  prep_kernel<<<(prep_n + 255) / 256, 256, 0, stream>>>(attn_W1, W_hh, W1T, WhhB);
  hproj_kernel<<<(BATCH * TLEN) / 16, 256, 0, stream>>>(H, attn_b1, W1T, HWb);
  decoder_kernel<<<BATCH / 16, 256, 0, stream>>>(H, Y, attn_W2, attn_b2, W_ih, b_ih,
                                                 b_hh, fc_W, fc_b, fcf_W, fcf_b, HWb,
                                                 W1T, WhhB, (float*)d_out);
}
